// SSMOpticalFlow_2173253452161
// MI455X (gfx1250) — matrix operand order repaired and checked by reading, not compiled, not run on hardware
//
#include <hip/hip_runtime.h>
#include <hip/hip_bf16.h>

#define NB 8  // batch

typedef __attribute__((ext_vector_type(16))) __bf16 v16bf;
typedef __attribute__((ext_vector_type(8)))  __bf16 v8bf;
typedef __attribute__((ext_vector_type(8)))  float  v8f;

static __device__ __forceinline__ int imin(int a, int b) { return a < b ? a : b; }
static __device__ __forceinline__ int imax(int a, int b) { return a > b ? a : b; }

static __device__ __forceinline__ float gelu_tanh(float x) {
  float x3 = x * x * x;
  return 0.5f * x * (1.f + tanhf(0.7978845608028654f * (x + 0.044715f * x3)));
}

// ---------------------------------------------------------------- utilities
__global__ void k_fill0(float* p, int n) {
  int i = blockIdx.x * blockDim.x + threadIdx.x;
  if (i < n) p[i] = 0.f;
}

__global__ void k_add(const float* __restrict__ a, const float* __restrict__ b,
                      float* __restrict__ dst, int n) {
  int i = blockIdx.x * blockDim.x + threadIdx.x;
  if (i < n) dst[i] = a[i] + b[i];
}

// align-corners separable bilinear resize, NCHW, with output scale multiplier
__global__ void k_resize_ac(const float* __restrict__ src, float* __restrict__ dst,
                            int C, int IH, int IW, int OH, int OW, float mult) {
  int i = blockIdx.x * blockDim.x + threadIdx.x;
  int total = NB * C * OH * OW;
  if (i >= total) return;
  int ox = i % OW; int t = i / OW; int oy = t % OH; t /= OH; int c = t % C; int b = t / C;
  int y0, y1, x0, x1; float wy, wx;
  if (OH == 1) { y0 = y1 = 0; wy = 0.f; }
  else {
    float sy = (float)oy * (float)(IH - 1) / (float)(OH - 1);
    y0 = (int)floorf(sy); y0 = imax(0, imin(y0, IH - 1));
    y1 = imin(y0 + 1, IH - 1); wy = sy - (float)y0;
  }
  if (OW == 1) { x0 = x1 = 0; wx = 0.f; }
  else {
    float sx = (float)ox * (float)(IW - 1) / (float)(OW - 1);
    x0 = (int)floorf(sx); x0 = imax(0, imin(x0, IW - 1));
    x1 = imin(x0 + 1, IW - 1); wx = sx - (float)x0;
  }
  const float* p = src + (size_t)(b * C + c) * IH * IW;
  float v00 = p[y0 * IW + x0], v01 = p[y0 * IW + x1];
  float v10 = p[y1 * IW + x0], v11 = p[y1 * IW + x1];
  dst[i] = mult * ((1.f - wy) * ((1.f - wx) * v00 + wx * v01) +
                   wy * ((1.f - wx) * v10 + wx * v11));
}

// channel-wise concat helper: copy src (NB,srcC,HW) into dst (NB,dstC,HW) at channel offset
__global__ void k_copy_ch(const float* __restrict__ src, float* __restrict__ dst,
                          int srcC, int dstC, int cOff, int HW) {
  int i = blockIdx.x * blockDim.x + threadIdx.x;
  int total = NB * srcC * HW;
  if (i >= total) return;
  int p = i % HW; int t = i / HW; int c = t % srcC; int b = t / srcC;
  dst[((size_t)(b * dstC + cOff + c)) * HW + p] = src[i];
}

// NCHW f32 -> halo-padded NHWC bf16: dst[b][(yy)*(W+2)+xx][c], c-stride Cs (>=Cin, mult of 8).
// Border, channel padding and the trailing slack are all written to 0 (WMMA propagates NaN,
// so every byte the conv kernel may touch must be a finite value).
__global__ void k_halo_bf16(const float* __restrict__ src, __bf16* __restrict__ dst,
                            int Cin, int Cs, int H, int W, int total) {
  int i = blockIdx.x * blockDim.x + threadIdx.x;
  if (i >= total) return;
  int NH = (H + 2) * (W + 2);
  int interior = NB * NH * Cs;
  __bf16 v = (__bf16)0.f;
  if (i < interior) {
    int c = i % Cs; int t = i / Cs; int px = t % NH; int b = t / NH;
    int yy = px / (W + 2), xx = px - (px / (W + 2)) * (W + 2);
    if (c < Cin && yy >= 1 && yy <= H && xx >= 1 && xx <= W)
      v = (__bf16)src[((size_t)(b * Cin + c)) * H * W + (yy - 1) * W + (xx - 1)];
  }
  dst[i] = v;
}

// pack conv weights [Cout][Cin][3][3] f32 -> [Mpad][Kpad] bf16, K = tap*Cinp + cin,
// tap = ky*3+kx, Cinp = ceil32(Cin). Padding rows/channels are zero.
__global__ void k_pack_w(const float* __restrict__ w, __bf16* __restrict__ wp,
                         int Cout, int Cin, int Cinp, int Mpad) {
  int Kpad = 9 * Cinp;
  int i = blockIdx.x * blockDim.x + threadIdx.x;
  if (i >= Mpad * Kpad) return;
  int k = i % Kpad; int m = i / Kpad;
  int tap = k / Cinp; int cin = k - tap * Cinp;
  int ky = tap / 3; int kx = tap - ky * 3;
  float v = 0.f;
  if (m < Cout && cin < Cin)
    v = w[(((size_t)m * Cin + cin) * 3 + ky) * 3 + kx];
  wp[i] = (__bf16)v;
}

// --------------------------------------------------- implicit-GEMM conv3x3 (bf16 WMMA)
// Per wave: 32 cout x 32 pixels via 2x2 tiles of v_wmma_f32_16x16x32_bf16.
// K loop: 9 taps x (Cinp/32) chunks; every fragment = two 16B global_load_b128,
// no bounds checks (halo-padded input), no divides in the loop.
__global__ __launch_bounds__(128) void k_conv3x3_wmma(
    const __bf16* __restrict__ inh,  // [NB][(H+2)*(W+2)][Cs] halo NHWC bf16
    const __bf16* __restrict__ wp,   // [Mpad][9*Cinp] bf16
    const float* __restrict__ bias,
    const float* __restrict__ bng, const float* __restrict__ bnb,
    const float* __restrict__ bnm, const float* __restrict__ bnv,
    float* __restrict__ out,         // NCHW f32
    int Cin, int Cs, int Cinp, int Cout, int H, int W, int relu) {
  int lane  = threadIdx.x & 31;
  int wave  = threadIdx.x >> 5;
  int half  = lane >> 4;    // K-half selector
  int lan16 = lane & 15;    // M row (A) / N pixel (B,D)
  int b  = blockIdx.z;
  int m0 = blockIdx.y * 32;
  int p0 = (blockIdx.x * 4 + wave) * 32;
  int HW = H * W;
  int W2 = W + 2;
  int Kpad = 9 * Cinp;

  int n0 = p0 + lan16, n1 = p0 + 16 + lan16;
  int nc0 = n0 < HW ? n0 : HW - 1;
  int nc1 = n1 < HW ? n1 : HW - 1;
  int y0 = nc0 / W, x0 = nc0 - (nc0 / W) * W;
  int y1 = nc1 / W, x1 = nc1 - (nc1 / W) * W;
  const __bf16* pb = inh + (size_t)b * (H + 2) * W2 * Cs;
  int base0 = ((y0 + 1) * W2 + (x0 + 1)) * Cs + 8 * half;
  int base1 = ((y1 + 1) * W2 + (x1 + 1)) * Cs + 8 * half;

  const __bf16* arow0 = wp + (size_t)(m0 + lan16) * Kpad + 8 * half;
  const __bf16* arow1 = arow0 + (size_t)16 * Kpad;

  v8f acc00 = {}, acc01 = {}, acc10 = {}, acc11 = {};
#pragma unroll
  for (int tap = 0; tap < 9; tap++) {
    int dy = tap / 3 - 1, dx = tap - (tap / 3) * 3 - 1;
    int toff = (dy * W2 + dx) * Cs;
    const __bf16* bp0 = pb + base0 + toff;
    const __bf16* bp1 = pb + base1 + toff;
    const __bf16* ap0 = arow0 + tap * Cinp;
    const __bf16* ap1 = arow1 + tap * Cinp;
    for (int cb = 0; cb < Cinp; cb += 32) {
      union { v16bf v; v8bf h[2]; } A0, A1, B0, B1;
      // A layout: e0..7 <- k = kb+8*half+e ; e8..15 <- k = kb+16+8*half+e
      A0.h[0] = *(const v8bf*)(ap0 + cb);
      A0.h[1] = *(const v8bf*)(ap0 + cb + 16);
      A1.h[0] = *(const v8bf*)(ap1 + cb);
      A1.h[1] = *(const v8bf*)(ap1 + cb + 16);
      // B layout: e0..7 <- k = kb+8*half+e ; e8..15 <- k = kb+16+8*half+e
      B0.h[0] = *(const v8bf*)(bp0 + cb);
      B0.h[1] = *(const v8bf*)(bp0 + cb + 16);
      B1.h[0] = *(const v8bf*)(bp1 + cb);
      B1.h[1] = *(const v8bf*)(bp1 + cb + 16);
      acc00 = __builtin_amdgcn_wmma_f32_16x16x32_bf16(false, A0.v, false, B0.v,
                                                      (short)0, acc00, false, false);
      acc01 = __builtin_amdgcn_wmma_f32_16x16x32_bf16(false, A0.v, false, B1.v,
                                                      (short)0, acc01, false, false);
      acc10 = __builtin_amdgcn_wmma_f32_16x16x32_bf16(false, A1.v, false, B0.v,
                                                      (short)0, acc10, false, false);
      acc11 = __builtin_amdgcn_wmma_f32_16x16x32_bf16(false, A1.v, false, B1.v,
                                                      (short)0, acc11, false, false);
    }
  }

  // epilogue: D layout -> M = r + 8*half, N = lan16; fused bias + BN + ReLU
#pragma unroll
  for (int mi = 0; mi < 2; mi++) {
#pragma unroll
    for (int r = 0; r < 8; r++) {
      int m = m0 + 16 * mi + r + 8 * half;
      if (m < Cout) {
        float s = 1.f, sh = bias[m];
        if (bng) {
          float inv = rsqrtf(bnv[m] + 1e-5f) * bng[m];
          s = inv; sh = (bias[m] - bnm[m]) * inv + bnb[m];
        }
        v8f a0 = mi ? acc10 : acc00;
        v8f a1 = mi ? acc11 : acc01;
        float v0 = a0[r] * s + sh;
        float v1 = a1[r] * s + sh;
        if (relu) { v0 = fmaxf(v0, 0.f); v1 = fmaxf(v1, 0.f); }
        float* orow = out + ((size_t)b * Cout + m) * HW;
        if (n0 < HW) orow[n0] = v0;
        if (n1 < HW) orow[n1] = v1;
      }
    }
  }
}

// ---------------------------------------------------------------- flow warp (bilinear)
__global__ void k_flow_warp(const float* __restrict__ feat, const float* __restrict__ flow,
                            float* __restrict__ out, int C, int H, int W) {
  int i = blockIdx.x * blockDim.x + threadIdx.x;
  int total = NB * C * H * W;
  if (i >= total) return;
  int x = i % W; int t = i / W; int y = t % H; t /= H; int c = t % C; int b = t / C;
  int HW = H * W;
  float fx = flow[((size_t)(b * 2 + 0)) * HW + y * W + x];
  float fy = flow[((size_t)(b * 2 + 1)) * HW + y * W + x];
  float X = (float)x + fx, Y = (float)y + fy;
  float x0f = floorf(X), y0f = floorf(Y);
  float wx = X - x0f, wy = Y - y0f;
  int x0 = (int)x0f, y0 = (int)y0f;
  const float* f = feat + ((size_t)(b * C + c)) * HW;
  float r = 0.f;
#pragma unroll
  for (int dy = 0; dy < 2; dy++)
#pragma unroll
    for (int dx = 0; dx < 2; dx++) {
      int xi = x0 + dx, yi = y0 + dy;
      if (xi >= 0 && xi < W && yi >= 0 && yi < H) {
        float wgt = (dy ? wy : 1.f - wy) * (dx ? wx : 1.f - wx);
        r += f[yi * W + xi] * wgt;
      }
    }
  out[i] = r;
}

// ------------------------------------------------ dilated correlation (R=4, DIL=2 -> 81 ch)
__global__ void k_dcorr(const float* __restrict__ f1, const float* __restrict__ f2,
                        float* __restrict__ out, int C, int H, int W) {
  int i = blockIdx.x * blockDim.x + threadIdx.x;
  int total = NB * 81 * H * W;
  if (i >= total) return;
  int x = i % W; int t = i / W; int y = t % H; t /= H; int o = t % 81; int b = t / 81;
  int HW = H * W;
  int dy = (o / 9 - 4) * 2, dx = (o % 9 - 4) * 2;
  int yy = y + dy, xx = x + dx;
  float s = 0.f;
  if (yy >= 0 && yy < H && xx >= 0 && xx < W) {
    const float* p1 = f1 + (size_t)b * C * HW + y * W + x;
    const float* p2 = f2 + (size_t)b * C * HW + yy * W + xx;
    for (int c = 0; c < C; c++) s += p1[(size_t)c * HW] * p2[(size_t)c * HW];
  }
  out[i] = s / (float)C;
}

// ---------------------------------------------------- S5 block (d=2, s=2), one thread/batch
__global__ void k_ssm(const float* __restrict__ xup, float* __restrict__ attn,
                      const float* __restrict__ Bm, const float* __restrict__ Cm,
                      const float* __restrict__ Dv, const float* __restrict__ Wd,
                      const float* __restrict__ We, const float* __restrict__ a_log,
                      const float* __restrict__ ln1b, const float* __restrict__ ln1g,
                      const float* __restrict__ ln2b, const float* __restrict__ ln2g,
                      const float* __restrict__ log_dt, int HW) {
  int b = blockIdx.x * blockDim.x + threadIdx.x;
  if (b >= NB) return;
  float dt0 = expf(log_dt[0]), dt1 = expf(log_dt[1]);
  float A0 = -expf(a_log[0]), A1 = -expf(a_log[1]);
  float Ad0 = expf(A0 * dt0), Ad1 = expf(A1 * dt1);
  float sc0 = (Ad0 - 1.f) / A0, sc1 = (Ad1 - 1.f) / A1;
  float Bd00 = sc0 * Bm[0], Bd01 = sc0 * Bm[1];   // Bd[s][d]
  float Bd10 = sc1 * Bm[2], Bd11 = sc1 * Bm[3];
  float g1a = ln1g[0], g1b = ln1g[1], b1a = ln1b[0], b1b = ln1b[1];
  float g2a = ln2g[0], g2b = ln2g[1], b2a = ln2b[0], b2b = ln2b[1];
  float C00 = Cm[0], C01 = Cm[1], C10 = Cm[2], C11 = Cm[3];  // C[d][s]
  float D0 = Dv[0], D1 = Dv[1];
  const float* xa = xup + (size_t)(b * 2 + 0) * HW;
  const float* xb = xup + (size_t)(b * 2 + 1) * HW;
  float* oa = attn + (size_t)(b * 2 + 0) * HW;
  float* ob = attn + (size_t)(b * 2 + 1) * HW;
  float h0 = 0.f, h1 = 0.f;
  for (int t = 0; t < HW; t++) {
    float u = xa[t], v = xb[t];
    float m = 0.5f * (u + v);
    float cu = u - m, cv = v - m;
    float inv = rsqrtf(0.5f * (cu * cu + cv * cv) + 1e-5f);
    float f0 = cu * inv * g1a + b1a;
    float f1 = cv * inv * g1b + b1b;
    h0 = Ad0 * h0 + f0 * Bd00 + f1 * Bd01;
    h1 = Ad1 * h1 + f0 * Bd10 + f1 * Bd11;
    float y0 = h0 * C00 + h1 * C01 + D0 * f0;
    float y1 = h0 * C10 + h1 * C11 + D1 * f1;
    float r0 = gelu_tanh(y0) + f0;
    float r1 = gelu_tanh(y1) + f1;
    float m2 = 0.5f * (r0 + r1);
    float d0 = r0 - m2, d1 = r1 - m2;
    float inv2 = rsqrtf(0.5f * (d0 * d0 + d1 * d1) + 1e-5f);
    float z0 = d0 * inv2 * g2a + b2a;
    float z1 = d1 * inv2 * g2b + b2b;
    float t0 = z0 * We[0] + z1 * We[1];
    float t1 = z0 * We[2] + z1 * We[3];
    float o0 = t0 * Wd[0] + t1 * Wd[1];
    float o1 = t0 * Wd[2] + t1 * Wd[3];
    oa[t] = o0 + z0;
    ob[t] = o1 + z1;
  }
}

// conf = sigmoid(dc[2]); out = conf*(x_up+delta) + (1-conf)*attn
__global__ void k_cdc_combine(const float* __restrict__ xup, const float* __restrict__ dc,
                              const float* __restrict__ attn, float* __restrict__ flow,
                              int HW) {
  int i = blockIdx.x * blockDim.x + threadIdx.x;
  int total = NB * 2 * HW;
  if (i >= total) return;
  int p = i % HW; int t = i / HW; int c = t % 2; int b = t / 2;
  float conf = 1.f / (1.f + expf(-dc[((size_t)(b * 3 + 2)) * HW + p]));
  float delta = dc[((size_t)(b * 3 + c)) * HW + p];
  float xu = xup[((size_t)(b * 2 + c)) * HW + p];
  float at = attn[((size_t)(b * 2 + c)) * HW + p];
  flow[i] = conf * (xu + delta) + (1.f - conf) * at;
}

// ============================================================= host orchestration
extern "C" void kernel_launch(void* const* d_in, const int* in_sizes, int n_in,
                              void* d_out, int out_size, void* d_ws, size_t ws_size,
                              hipStream_t stream) {
  (void)in_sizes; (void)n_in; (void)out_size; (void)ws_size;

  // jax pytree flattening: dicts iterate in sorted-key order.
  // top: curr_c1..c4, params, prev_c1..c3
  // params: dec1, dec2, dec3, up1, up2, up3
  // dec: bn0{b,g,m,v} bn1{..} c0{b,w} c1{b,w} c2{b,w}                (14 leaves)
  // cdc: bn0 bn1 c0 c1 c2 ssm{B,C,D,Wd,We,a_log,ln1b,ln1g,ln2b,ln2g,log_dt} (25)
  const int I_CURR_C1 = 0, I_CURR_C2 = 1, I_CURR_C3 = 2;
  const int D1 = 4, D2 = 18, D3 = 32, U1 = 46, U2 = 71, U3 = 96;
  const int I_PREV_C1 = 121, I_PREV_C2 = 122, I_PREV_C3 = 123;
  auto F = [&](int i) { return (const float*)d_in[i]; };

  char* cur = (char*)d_ws;
  auto alloc = [&](size_t bytes) -> void* {
    void* r = cur; cur += (bytes + 255) & ~(size_t)255; return r;
  };
  auto cdiv = [](int a, int b) { return (a + b - 1) / b; };

  // persistent flow ping-pong (sized for the largest scale, 80x80)
  float* flowA = (float*)alloc((size_t)NB * 2 * 6400 * 4);
  float* flowB = (float*)alloc((size_t)NB * 2 * 6400 * 4);

  // conv3x3 + optional BN + optional ReLU via bf16 WMMA implicit GEMM
  auto run_conv = [&](const float* in_nchw, int Cin, int Cout, int H, int W,
                      const float* w, const float* bias,
                      const float* bng, const float* bnb, const float* bnm,
                      const float* bnv, int relu, float* out_nchw) {
    int HW = H * W;
    int Cinp = (Cin + 31) & ~31;       // K-chunk padding
    int Cs   = (Cin + 7) & ~7;         // pixel stride (16B-aligned runs)
    int Kpad = 9 * Cinp;
    int Mpad = (Cout + 31) & ~31;      // 2 M-tiles per wave
    int NH = (H + 2) * (W + 2);
    int htot = NB * NH * Cs + 64;      // +slack (zeroed) for cin-run overshoot
    __bf16* wp  = (__bf16*)alloc((size_t)Mpad * Kpad * 2);
    __bf16* inh = (__bf16*)alloc((size_t)htot * 2);
    k_pack_w<<<cdiv(Mpad * Kpad, 256), 256, 0, stream>>>(w, wp, Cout, Cin, Cinp, Mpad);
    k_halo_bf16<<<cdiv(htot, 256), 256, 0, stream>>>(in_nchw, inh, Cin, Cs, H, W, htot);
    dim3 grid(cdiv(HW, 128), Mpad / 32, NB);
    k_conv3x3_wmma<<<grid, 128, 0, stream>>>(
        inh, wp, bias, bng, bnb, bnm, bnv, out_nchw, Cin, Cs, Cinp, Cout, H, W, relu);
  };

  // one pyramid level: flow_up (cdc) + decoder
  auto do_scale = [&](const float* skip, int C, int H, int W, int Hc, int Wc,
                      const float* prevf, int U, int D, int hidUp, int hidDec,
                      float* flowIn, float* flowOut) {
    char* save = cur;
    int HW = H * W;
    // --- flow_up: f = resize(flow, H, W) * 2
    float* f2 = (float*)alloc((size_t)NB * 2 * HW * 4);
    k_resize_ac<<<cdiv(NB * 2 * HW, 256), 256, 0, stream>>>(flowIn, f2, 2, Hc, Wc, H, W, 2.0f);
    // --- cdc: x_up = resize(resize(f, 2H, 2W), H, W)
    float* xupa = (float*)alloc((size_t)NB * 2 * 4 * HW * 4);
    k_resize_ac<<<cdiv(NB * 8 * HW, 256), 256, 0, stream>>>(f2, xupa, 2, H, W, 2 * H, 2 * W, 1.0f);
    float* xup = (float*)alloc((size_t)NB * 2 * HW * 4);
    k_resize_ac<<<cdiv(NB * 2 * HW, 256), 256, 0, stream>>>(xupa, xup, 2, 2 * H, 2 * W, H, W, 1.0f);
    // fused = concat(x_up, skip)
    int Cf = 2 + C;
    float* fused = (float*)alloc((size_t)NB * Cf * HW * 4);
    k_copy_ch<<<cdiv(NB * 2 * HW, 256), 256, 0, stream>>>(xup, fused, 2, Cf, 0, HW);
    k_copy_ch<<<cdiv(NB * C * HW, 256), 256, 0, stream>>>(skip, fused, C, Cf, 2, HW);
    float* h0 = (float*)alloc((size_t)NB * hidUp * HW * 4);
    run_conv(fused, Cf, hidUp, H, W, F(U + 9), F(U + 8),
             F(U + 1), F(U + 0), F(U + 2), F(U + 3), 1, h0);
    float* h1 = (float*)alloc((size_t)NB * hidUp * HW * 4);
    run_conv(h0, hidUp, hidUp, H, W, F(U + 11), F(U + 10),
             F(U + 5), F(U + 4), F(U + 6), F(U + 7), 1, h1);
    float* dc = (float*)alloc((size_t)NB * 3 * HW * 4);
    run_conv(h1, hidUp, 3, H, W, F(U + 13), F(U + 12),
             nullptr, nullptr, nullptr, nullptr, 0, dc);
    float* attn = (float*)alloc((size_t)NB * 2 * HW * 4);
    int S = U + 14;  // ssm leaves: B,C,D,Wd,We,a_log,ln1b,ln1g,ln2b,ln2g,log_dt
    k_ssm<<<1, 32, 0, stream>>>(xup, attn, F(S + 0), F(S + 1), F(S + 2), F(S + 3),
                                F(S + 4), F(S + 5), F(S + 6), F(S + 7), F(S + 8),
                                F(S + 9), F(S + 10), HW);
    k_cdc_combine<<<cdiv(NB * 2 * HW, 256), 256, 0, stream>>>(xup, dc, attn, flowOut, HW);
    cur = save;  // release cdc scratch
    // --- decoder
    float* warped = (float*)alloc((size_t)NB * C * HW * 4);
    k_flow_warp<<<cdiv(NB * C * HW, 256), 256, 0, stream>>>(prevf, flowOut, warped, C, H, W);
    float* cost = (float*)alloc((size_t)NB * 81 * HW * 4);
    k_dcorr<<<cdiv(NB * 81 * HW, 256), 256, 0, stream>>>(skip, warped, cost, C, H, W);
    int Ci = 83 + C;
    float* inp = (float*)alloc((size_t)NB * Ci * HW * 4);
    k_copy_ch<<<cdiv(NB * 81 * HW, 256), 256, 0, stream>>>(cost, inp, 81, Ci, 0, HW);
    k_copy_ch<<<cdiv(NB * C * HW, 256), 256, 0, stream>>>(skip, inp, C, Ci, 81, HW);
    k_copy_ch<<<cdiv(NB * 2 * HW, 256), 256, 0, stream>>>(flowOut, inp, 2, Ci, 81 + C, HW);
    float* e0 = (float*)alloc((size_t)NB * hidDec * HW * 4);
    run_conv(inp, Ci, hidDec, H, W, F(D + 9), F(D + 8),
             F(D + 1), F(D + 0), F(D + 2), F(D + 3), 1, e0);
    float* e1 = (float*)alloc((size_t)NB * hidDec * HW * 4);
    run_conv(e0, hidDec, hidDec, H, W, F(D + 11), F(D + 10),
             F(D + 5), F(D + 4), F(D + 6), F(D + 7), 1, e1);
    float* delta = (float*)alloc((size_t)NB * 2 * HW * 4);
    run_conv(e1, hidDec, 2, H, W, F(D + 13), F(D + 12),
             nullptr, nullptr, nullptr, nullptr, 0, delta);
    k_add<<<cdiv(NB * 2 * HW, 256), 256, 0, stream>>>(flowOut, delta, flowOut, NB * 2 * HW);
    cur = save;  // release decoder scratch
  };

  // flow_s4 = zeros (B,2,10,10)
  k_fill0<<<cdiv(NB * 2 * 100, 256), 256, 0, stream>>>(flowA, NB * 2 * 100);

  // scale 3: 20x20, C=256, up3 hidden 128, dec3 hidden 256
  do_scale(F(I_CURR_C3), 256, 20, 20, 10, 10, F(I_PREV_C3), U3, D3, 128, 256, flowA, flowB);
  // scale 2: 40x40, C=128, up2 hidden 64, dec2 hidden 128
  do_scale(F(I_CURR_C2), 128, 40, 40, 20, 20, F(I_PREV_C2), U2, D2, 64, 128, flowB, flowA);
  // scale 1: 80x80, C=64, up1 hidden 32, dec1 hidden 64
  do_scale(F(I_CURR_C1), 64, 80, 80, 40, 40, F(I_PREV_C1), U1, D1, 32, 64, flowA, flowB);

  // final: resize to 320x320, *4
  k_resize_ac<<<cdiv(NB * 2 * 320 * 320, 256), 256, 0, stream>>>(
      flowB, (float*)d_out, 2, 80, 80, 320, 320, 4.0f);
}
